// Attention_21406117003357
// MI455X (gfx1250) — compile-verified, matrix operand order checked by reading, not run on hardware
//
#include <hip/hip_runtime.h>
#include <math.h>

typedef __attribute__((ext_vector_type(16))) __bf16 v16bf;
typedef __attribute__((ext_vector_type(8)))  float  v8f;

#define B_ 128
#define N_ 196
#define C_ 2048
#define H_ 512
#define A_ 512
#define R_ (B_*N_)     // 25088 rows of the fused (B*N, C) matrix
#define KT_ (C_/32)    // 64 K-steps of 32
#define AT_ (A_/16)    // 32 A-tiles of 16

// ---------------------------------------------------------------------------
// Repack W_img (C,A) fp32 row-major -> bf16 in WMMA B-matrix register layout.
// Chunk t (32B) = ((a_tile*KT_ + k_tile)*32 + lane)*16 bf16 elems.
// Lane holds column a = a_tile*16 + (lane&15); K pattern per ISA 16-bit layout:
//   elems 0..7 -> K = k0+0..7, elems 8..15 -> K = k0+16..23,
//   k0 = k_tile*32 + (lane>=16 ? 8 : 0).
// ---------------------------------------------------------------------------
__global__ __launch_bounds__(256) void pack_w_kernel(const float* __restrict__ Wimg,
                                                     v16bf* __restrict__ wsB) {
    int t    = blockIdx.x * 256 + threadIdx.x;    // 65536 chunks total
    int lane = t & 31;
    int kt   = (t >> 5) & (KT_ - 1);
    int at   = t >> 11;
    int k0   = kt * 32 + ((lane & 16) ? 8 : 0);
    int a    = at * 16 + (lane & 15);
    v16bf frag;
#pragma unroll
    for (int e = 0; e < 8; ++e)
        frag[e] = (__bf16)Wimg[(k0 + e) * A_ + a];
#pragma unroll
    for (int e = 0; e < 8; ++e)
        frag[8 + e] = (__bf16)Wimg[(k0 + 16 + e) * A_ + a];
    wsB[t] = frag;
}

// ---------------------------------------------------------------------------
// hid_att_eff[b,a] = hidden[b,:] @ W_hid[:,a] + b_hid[a] + b_img[a]
// beta[b]          = sigmoid(hidden[b,:] @ w_beta + b_beta)
// ---------------------------------------------------------------------------
__global__ __launch_bounds__(256) void hid_kernel(const float* __restrict__ hidden,
                                                  const float* __restrict__ Whid,
                                                  const float* __restrict__ bhid,
                                                  const float* __restrict__ bimg,
                                                  const float* __restrict__ wbeta,
                                                  const float* __restrict__ bbeta,
                                                  float* __restrict__ hidatt,
                                                  float* __restrict__ beta) {
    __shared__ float lh[H_];
    __shared__ float red[256];
    const int b = blockIdx.x, t = threadIdx.x;
    lh[t]       = hidden[b * H_ + t];
    lh[t + 256] = hidden[b * H_ + t + 256];
    __syncthreads();
#pragma unroll
    for (int rep = 0; rep < 2; ++rep) {
        int a = t + rep * 256;
        float acc = bhid[a] + bimg[a];
        for (int h = 0; h < H_; ++h)
            acc = fmaf(lh[h], Whid[h * A_ + a], acc);
        hidatt[b * A_ + a] = acc;
    }
    float p = fmaf(lh[t], wbeta[t], lh[t + 256] * wbeta[t + 256]);
    red[t] = p;
    __syncthreads();
    for (int off = 128; off > 0; off >>= 1) {
        if (t < off) red[t] += red[t + off];
        __syncthreads();
    }
    if (t == 0) {
        float x = red[0] + bbeta[0];
        beta[b] = 1.0f / (1.0f + __expf(-x));
    }
}

// ---------------------------------------------------------------------------
// Fused: scores[r] = sum_a relu((img@W_img)[r,a] + hid_att_eff[b(r),a]) * w_att[a]
// (b_att dropped: softmax is shift-invariant.)
// Block = 16 rows x full A. 8 waves x 4 A-tiles. K-loop unrolled by 2 with
// two ping-pong fragment buffers: loads write each buffer directly (no
// rotate-copies), WMMAs of one buffer overlap loads of the other.
// ---------------------------------------------------------------------------
__global__ __launch_bounds__(256) void score_kernel(const float* __restrict__ img,
                                                    const v16bf* __restrict__ wsB,
                                                    const float* __restrict__ hidatt,
                                                    const float* __restrict__ watt,
                                                    float* __restrict__ scores) {
    __shared__ float lds_p[16 * 8];
    const int t        = threadIdx.x;
    const int wave     = t >> 5;
    const int lane     = t & 31;
    const int m16      = lane & 15;
    const int half     = lane >> 4;          // 0: rows 0..7, 1: rows 8..15
    const int row_base = blockIdx.x * 16;    // R_ = 1568*16 exactly

    const float* pa = img + (size_t)(row_base + m16) * C_ + (half ? 8 : 0);
    const v16bf* pb = wsB + (size_t)(wave * 4) * KT_ * 32 + lane;  // + j*KT_*32 + kt*32

    auto loadA = [&](int kt) -> v16bf {
        const float* p = pa + kt * 32;
        float4 l0 = *(const float4*)(p);
        float4 l1 = *(const float4*)(p + 4);
        float4 l2 = *(const float4*)(p + 16);
        float4 l3 = *(const float4*)(p + 20);
        v16bf af;
        af[0]  = (__bf16)l0.x; af[1]  = (__bf16)l0.y; af[2]  = (__bf16)l0.z; af[3]  = (__bf16)l0.w;
        af[4]  = (__bf16)l1.x; af[5]  = (__bf16)l1.y; af[6]  = (__bf16)l1.z; af[7]  = (__bf16)l1.w;
        af[8]  = (__bf16)l2.x; af[9]  = (__bf16)l2.y; af[10] = (__bf16)l2.z; af[11] = (__bf16)l2.w;
        af[12] = (__bf16)l3.x; af[13] = (__bf16)l3.y; af[14] = (__bf16)l3.z; af[15] = (__bf16)l3.w;
        return af;
    };

    v8f acc[4];
#pragma unroll
    for (int j = 0; j < 4; ++j)
#pragma unroll
        for (int i = 0; i < 8; ++i) acc[j][i] = 0.0f;

    // buffer set A: even kt; buffer set B: odd kt
    v16bf afA, afB;
    v16bf bfA[4], bfB[4];

    afA = loadA(0);
#pragma unroll
    for (int j = 0; j < 4; ++j) bfA[j] = pb[(size_t)j * KT_ * 32];

    for (int kt = 0; kt < KT_ - 2; kt += 2) {
        __builtin_prefetch(pa + (kt + 16) * 32, 0, 1);   // global_prefetch ahead
        // load odd buffer, run even WMMAs
        afB = loadA(kt + 1);
#pragma unroll
        for (int j = 0; j < 4; ++j) bfB[j] = pb[(size_t)j * KT_ * 32 + (kt + 1) * 32];
#pragma unroll
        for (int j = 0; j < 4; ++j)
            acc[j] = __builtin_amdgcn_wmma_f32_16x16x32_bf16(
                false, afA, false, bfA[j], (short)0, acc[j], false, false);
        // load even buffer for kt+2, run odd WMMAs
        afA = loadA(kt + 2);
#pragma unroll
        for (int j = 0; j < 4; ++j) bfA[j] = pb[(size_t)j * KT_ * 32 + (kt + 2) * 32];
#pragma unroll
        for (int j = 0; j < 4; ++j)
            acc[j] = __builtin_amdgcn_wmma_f32_16x16x32_bf16(
                false, afB, false, bfB[j], (short)0, acc[j], false, false);
    }
    // tail: kt = KT_-2 (in buffer A), kt = KT_-1 (load into buffer B)
    afB = loadA(KT_ - 1);
#pragma unroll
    for (int j = 0; j < 4; ++j) bfB[j] = pb[(size_t)j * KT_ * 32 + (KT_ - 1) * 32];
#pragma unroll
    for (int j = 0; j < 4; ++j)
        acc[j] = __builtin_amdgcn_wmma_f32_16x16x32_bf16(
            false, afA, false, bfA[j], (short)0, acc[j], false, false);
#pragma unroll
    for (int j = 0; j < 4; ++j)
        acc[j] = __builtin_amdgcn_wmma_f32_16x16x32_bf16(
            false, afB, false, bfB[j], (short)0, acc[j], false, false);

    // Epilogue: relu(acc + hid_att) * w_att, reduce over A and across lanes.
#pragma unroll
    for (int i = 0; i < 8; ++i) {
        int m  = i + half * 8;
        int r  = row_base + m;
        int bb = r / N_;
        float s = 0.0f;
#pragma unroll
        for (int j = 0; j < 4; ++j) {
            int a   = (wave * 4 + j) * 16 + m16;
            float v = acc[j][i] + hidatt[bb * A_ + a];
            v = fmaxf(v, 0.0f);
            s = fmaf(v, watt[a], s);
        }
        s += __shfl_xor(s, 1, 32);
        s += __shfl_xor(s, 2, 32);
        s += __shfl_xor(s, 4, 32);
        s += __shfl_xor(s, 8, 32);
        if (m16 == 0) lds_p[m * 8 + wave] = s;
    }
    __syncthreads();
    if (t < 16) {
        float s = 0.0f;
#pragma unroll
        for (int w = 0; w < 8; ++w) s += lds_p[t * 8 + w];
        scores[row_base + t] = s;
    }
}

// ---------------------------------------------------------------------------
// Per-batch softmax over N=196; weights written straight to d_out.
// ---------------------------------------------------------------------------
__global__ __launch_bounds__(256) void softmax_kernel(const float* __restrict__ scores,
                                                      float* __restrict__ wout) {
    __shared__ float red[256];
    const int b = blockIdx.x, t = threadIdx.x;
    float s = (t < N_) ? scores[b * N_ + t] : -3.0e38f;
    red[t] = s;
    __syncthreads();
    for (int off = 128; off > 0; off >>= 1) {
        if (t < off) red[t] = fmaxf(red[t], red[t + off]);
        __syncthreads();
    }
    float mx = red[0];
    __syncthreads();
    float e = (t < N_) ? __expf(s - mx) : 0.0f;
    red[t] = e;
    __syncthreads();
    for (int off = 128; off > 0; off >>= 1) {
        if (t < off) red[t] += red[t + off];
        __syncthreads();
    }
    if (t < N_) wout[b * N_ + t] = e * (1.0f / red[0]);
}

// ---------------------------------------------------------------------------
// context[b,:] = beta[b] * (weights[b,:]^T @ img[b]).  Grid = B*8 blocks;
// each block owns 256 channels (64 float4). 4 thread-groups partition N,
// LDS tree-reduce (fixed order -> deterministic). 1 KB contiguous per (n,g).
// ---------------------------------------------------------------------------
__global__ __launch_bounds__(256) void ctx_kernel(const float* __restrict__ img,
                                                  const float* __restrict__ wts_g,
                                                  const float* __restrict__ beta,
                                                  float* __restrict__ ctx) {
    __shared__ float wts[N_];
    __shared__ float4 part[256];
    const int blk = blockIdx.x;
    const int b = blk >> 3, chunk = blk & 7;
    const int t = threadIdx.x;

    if (t < N_) wts[t] = wts_g[b * N_ + t];
    __syncthreads();

    const int col = t & 63;            // float4 column within chunk
    const int g   = t >> 6;            // 0..3 N-partition
    const int vc  = chunk * 64 + col;  // global float4 column
    const float4* p0 = (const float4*)(img + (size_t)b * N_ * C_);

    float4 acc = make_float4(0.f, 0.f, 0.f, 0.f);
    for (int n = g; n < N_; n += 4) {
        float wn = wts[n];
        float4 v = p0[(size_t)n * (C_ / 4) + vc];
        acc.x = fmaf(wn, v.x, acc.x);
        acc.y = fmaf(wn, v.y, acc.y);
        acc.z = fmaf(wn, v.z, acc.z);
        acc.w = fmaf(wn, v.w, acc.w);
    }
    part[t] = acc;
    __syncthreads();
    if (g == 0) {
        float4 r = part[col];
#pragma unroll
        for (int gg = 1; gg < 4; ++gg) {
            float4 o = part[gg * 64 + col];
            r.x += o.x; r.y += o.y; r.z += o.z; r.w += o.w;
        }
        const float bb = beta[b];
        r.x *= bb; r.y *= bb; r.z *= bb; r.w *= bb;
        ((float4*)ctx)[(size_t)b * (C_ / 4) + vc] = r;
    }
}

// ---------------------------------------------------------------------------
extern "C" void kernel_launch(void* const* d_in, const int* in_sizes, int n_in,
                              void* d_out, int out_size, void* d_ws, size_t ws_size,
                              hipStream_t stream) {
    (void)in_sizes; (void)n_in; (void)out_size; (void)ws_size;
    const float* img    = (const float*)d_in[0];
    const float* hidden = (const float*)d_in[1];
    const float* Wimg   = (const float*)d_in[2];
    const float* bimg   = (const float*)d_in[3];
    const float* Whid   = (const float*)d_in[4];
    const float* bhid   = (const float*)d_in[5];
    const float* watt   = (const float*)d_in[6];
    // d_in[7] = b_att: uniform score shift, cancelled by softmax -> unused.
    const float* wbeta  = (const float*)d_in[8];
    const float* bbeta  = (const float*)d_in[9];

    char* ws = (char*)d_ws;
    v16bf* wsB    = (v16bf*)ws;                                   // 2 MB packed W_img bf16
    float* hidatt = (float*)(ws + (2u << 20));                    // 256 KB
    float* beta   = (float*)(ws + (2u << 20) + (256u << 10));     // 512 B (+pad)
    float* scores = (float*)(ws + (2u << 20) + (256u << 10) + 4096);

    float* ctx_out = (float*)d_out;                               // (B, C)
    float* w_out   = (float*)d_out + (size_t)B_ * C_;             // (B, N)

    pack_w_kernel<<<(AT_ * KT_ * 32) / 256, 256, 0, stream>>>(Wimg, wsB);
    hid_kernel<<<B_, 256, 0, stream>>>(hidden, Whid, bhid, bimg, wbeta, bbeta, hidatt, beta);
    score_kernel<<<R_ / 16, 256, 0, stream>>>(img, wsB, hidatt, watt, scores);
    softmax_kernel<<<B_, 256, 0, stream>>>(scores, w_out);
    ctx_kernel<<<B_ * 8, 256, 0, stream>>>(img, w_out, beta, ctx_out);
}
